// MotionInversionAdapter_33835752358445
// MI455X (gfx1250) — matrix operand order repaired and checked by reading, not compiled, not run on hardware
//
#include <hip/hip_runtime.h>
#include <hip/hip_bf16.h>

// ---------------------------------------------------------------------------
// MotionInversionAdapter attention block for MI455X (gfx1250, wave32, WMMA)
// ---------------------------------------------------------------------------

typedef __attribute__((ext_vector_type(16))) _Float16 v16h;
typedef __attribute__((ext_vector_type(8)))  _Float16 v8h;
typedef __attribute__((ext_vector_type(4)))  _Float16 v4h;
typedef __attribute__((ext_vector_type(8)))  float    v8f;

union V16 { v16h v; v8h h[2]; };

static constexpr int NUM_HEADS = 12;
static constexpr int HEAD_DIM  = 128;
static constexpr int DIM       = 1536;   // NUM_HEADS * HEAD_DIM
static constexpr int FRAMES    = 8;
static constexpr int SPATIAL   = 390;    // 15*26
static constexpr int SEQ       = 3120;   // FRAMES*SPATIAL
static constexpr int SEQP      = 3136;   // padded to 32-key chunks (98*32)
static constexpr int MTILES    = SEQ / 16;       // 195
static constexpr int KSTEPS    = DIM / 32;       // 48
static constexpr float SM_SCALE = 0.08838834764831845f; // 1/sqrt(128)
static constexpr float EPS = 1e-5f;

__device__ __forceinline__ v8f wmma_f16(v16h a, v16h b, v8f c) {
    // D = A(16x32 f16) * B(32x16 f16) + C(16x16 f32)
    return __builtin_amdgcn_wmma_f32_16x16x32_f16(
        /*neg_a=*/false, a, /*neg_b=*/false, b,
        /*c_mod=*/(short)0, c, /*reuse_a=*/false, /*reuse_b=*/false);
}

// ---------------------------------------------------------------------------
// Kernel 0: f32 -> f16 conversion (vectorized x4)
// ---------------------------------------------------------------------------
__global__ void cvt_f32_f16(const float* __restrict__ in, _Float16* __restrict__ out, int n4) {
    int i = blockIdx.x * blockDim.x + threadIdx.x;
    if (i < n4) {
        float4 f = reinterpret_cast<const float4*>(in)[i];
        v4h h = { (_Float16)f.x, (_Float16)f.y, (_Float16)f.z, (_Float16)f.w };
        reinterpret_cast<v4h*>(out)[i] = h;
    }
}

// ---------------------------------------------------------------------------
// Kernel 1: GEMM (16-row tile x full 1536 cols) with fused epilogues.
//   mode 0: Q  = rope(rmsnorm(x@Wq+bq, gq)) + frame_bias  -> q_t  [h][128][SEQ]  f16
//   mode 1: K  = rope(rmsnorm(x@Wk+bk, gk)) + frame_bias  -> k_n  [h][SEQP][128] f16
//   mode 2: V  = x@Wv+bv + frame_bias + spatial_bias      -> v_n  [h][SEQP][128] f16
//   mode 3: out= attn@Wo + bo                             -> out_f [SEQ][DIM]    f32
// 256 threads = 8 waves; wave w owns N-tiles [w*12, w*12+12).
// ---------------------------------------------------------------------------
__global__ __launch_bounds__(256) void gemm_fused(
    const _Float16* __restrict__ A,      // [SEQ][DIM] f16
    const _Float16* __restrict__ W,      // [DIM][DIM] f16 (row-major K x N)
    const float*    __restrict__ bias,   // [DIM]
    const float*    __restrict__ gain,   // [DIM]      (modes 0/1)
    const float*    __restrict__ freqs,  // [SEQ][64]  (modes 0/1)
    const float*    __restrict__ fbias,  // [FRAMES][DIM]
    const float*    __restrict__ sbias,  // [SPATIAL][DIM] (mode 2)
    _Float16*       __restrict__ out_h,  // f16 dest (modes 0/1/2)
    float*          __restrict__ out_f,  // f32 dest (mode 3)
    int mode)
{
    extern __shared__ float smem[];      // modes 0/1: [16][1536] tile + [16] sumsq

    const int tid  = threadIdx.x;
    const int lane = tid & 31;
    const int wave = tid >> 5;
    const int hi   = (lane >> 4) & 1;    // lane half
    const int l15  = lane & 15;
    const int mt   = blockIdx.x;         // 16-row tile index
    const int arow = mt * 16 + l15;      // A-fragment row for this lane

    v8f acc[12];
#pragma unroll
    for (int i = 0; i < 12; ++i) acc[i] = v8f{};

    // ---- main K loop: 48 x (12 WMMAs) ----
    for (int kt = 0; kt < KSTEPS; ++kt) {
        V16 a;
        const _Float16* ap = A + (size_t)arow * DIM + kt * 32 + 8 * hi;
        a.h[0] = *reinterpret_cast<const v8h*>(ap);        // K = base .. base+7
        a.h[1] = *reinterpret_cast<const v8h*>(ap + 16);   // K = base+16 .. base+23

        const _Float16* wp = W + (size_t)(kt * 32) * DIM + wave * 192 + l15;
#pragma unroll
        for (int i = 0; i < 12; ++i) {
            V16 b;
#pragma unroll
            for (int s = 0; s < 16; ++s)
                b.v[s] = wp[(size_t)((s & 7) + 8 * hi + 16 * (s >> 3)) * DIM + i * 16];
            acc[i] = wmma_f16(a.v, b.v, acc[i]);
        }
    }

    // ---- epilogues ----
    if (mode == 2) {            // V projection
#pragma unroll
        for (int i = 0; i < 12; ++i) {
            const int n = wave * 192 + i * 16 + l15;
#pragma unroll
            for (int r = 0; r < 8; ++r) {
                const int s = mt * 16 + r + 8 * hi;
                float val = acc[i][r] + bias[n]
                          + fbias[(s / SPATIAL) * DIM + n]
                          + sbias[(s % SPATIAL) * DIM + n];
                const int head = n >> 7, ch = n & 127;
                out_h[((size_t)head * SEQP + s) * HEAD_DIM + ch] = (_Float16)val;
            }
        }
        return;
    }
    if (mode == 3) {            // output projection -> f32
#pragma unroll
        for (int i = 0; i < 12; ++i) {
            const int n = wave * 192 + i * 16 + l15;
#pragma unroll
            for (int r = 0; r < 8; ++r) {
                const int s = mt * 16 + r + 8 * hi;
                out_f[(size_t)s * DIM + n] = acc[i][r] + bias[n];
            }
        }
        return;
    }

    // modes 0/1: RMSNorm + RoPE + frame bias (needs the full 1536-wide row)
    float* tile = smem;                 // [16][1536]
    float* sq   = smem + 16 * DIM;      // [16]
    if (tid < 16) sq[tid] = 0.0f;
#pragma unroll
    for (int i = 0; i < 12; ++i) {
        const int n = wave * 192 + i * 16 + l15;
#pragma unroll
        for (int r = 0; r < 8; ++r) {
            const int rl = r + 8 * hi;
            tile[rl * DIM + n] = acc[i][r] + bias[n];
        }
    }
    __syncthreads();

    {   // per-row sum of squares: 16 threads per row, ds_add_f32 reduce
        const int r  = tid >> 4;
        const int c0 = tid & 15;
        float ssum = 0.0f;
        for (int c = c0; c < DIM; c += 16) {
            float x = tile[r * DIM + c];
            ssum += x * x;
        }
        atomicAdd(&sq[r], ssum);
    }
    __syncthreads();

    // RoPE over channel pairs: 16 rows x 768 pairs
    for (int it = tid; it < 16 * (DIM / 2); it += 256) {
        const int r = it / (DIM / 2);
        const int d = it % (DIM / 2);
        const int head = d >> 6;
        const int p    = d & 63;
        const int n0   = head * HEAD_DIM + 2 * p;
        const int s    = mt * 16 + r;
        const float inv = rsqrtf(sq[r] * (1.0f / (float)DIM) + EPS);
        const float xr = tile[r * DIM + n0]     * inv * gain[n0];
        const float xi = tile[r * DIM + n0 + 1] * inv * gain[n0 + 1];
        const float fr = freqs[s * 64 + p];
        const float cs = __cosf(fr), sn = __sinf(fr);
        const float* fb = fbias + (s / SPATIAL) * DIM;
        const float outr = xr * cs - xi * sn + fb[n0];
        const float outi = xr * sn + xi * cs + fb[n0 + 1];
        if (mode == 0) {        // Q transposed: [channel][SEQ]
            out_h[(size_t)n0 * SEQ + s]       = (_Float16)outr;
            out_h[(size_t)(n0 + 1) * SEQ + s] = (_Float16)outi;
        } else {                // K natural (padded): [h][SEQP][128]
            out_h[((size_t)head * SEQP + s) * HEAD_DIM + 2 * p]     = (_Float16)outr;
            out_h[((size_t)head * SEQP + s) * HEAD_DIM + 2 * p + 1] = (_Float16)outi;
        }
    }
}

// ---------------------------------------------------------------------------
// Kernel 2: flash attention, one wave per (head, 16-query tile).
// Computes S^T = K(16x128) @ Q^T(128x16) so the S^T C-layout (lane=q-row,
// reg=key) directly forms the A-fragment of P for the P@V WMMA — no LDS
// transpose. Softmax stats per lane, combined across lane halves via shfl_xor.
// ---------------------------------------------------------------------------
__global__ __launch_bounds__(128) void flash_attn(
    const _Float16* __restrict__ qT,   // [h][128][SEQ]
    const _Float16* __restrict__ kN,   // [h][SEQP][128]
    const _Float16* __restrict__ vN,   // [h][SEQP][128]
    _Float16*       __restrict__ attn) // [SEQ][DIM]
{
    const int lane = threadIdx.x & 31;
    const int wave = threadIdx.x >> 5;
    const int qt   = blockIdx.x * 4 + wave;
    const int head = blockIdx.y;
    if (qt >= MTILES) return;          // wave-uniform exit; EXEC stays all-1s

    const int hi  = (lane >> 4) & 1;
    const int l15 = lane & 15;

    const _Float16* q = qT + (size_t)head * HEAD_DIM * SEQ;
    const _Float16* k = kN + (size_t)head * SEQP * HEAD_DIM;
    const _Float16* v = vN + (size_t)head * SEQP * HEAD_DIM;

    // Q^T B-fragments (fixed for this q-tile): lane%16 = query column,
    // elements = contraction rows in table order (8*hi, 16+8*hi).
    V16 bq[4];
#pragma unroll
    for (int kc = 0; kc < 4; ++kc) {
        const _Float16* qp = q + (size_t)(kc * 32) * SEQ + qt * 16 + l15;
#pragma unroll
        for (int s = 0; s < 16; ++s)
            bq[kc].v[s] = qp[(size_t)((s & 7) + 8 * hi + 16 * (s >> 3)) * SEQ];
    }

    float m = -1e30f, l = 0.0f;
    v8f o[8];
#pragma unroll
    for (int ct = 0; ct < 8; ++ct) o[ct] = v8f{};

    for (int c0 = 0; c0 < SEQP; c0 += 32) {
        // ---- S^T tiles: t0 = keys c0..c0+15, t1 = keys c0+16..c0+31 ----
        v8f s0 = v8f{}, s1 = v8f{};
#pragma unroll
        for (int kc = 0; kc < 4; ++kc) {
            const int hb = kc * 32 + 8 * hi;
            V16 ka0, ka1;
            const _Float16* k0p = k + (size_t)(c0 + l15) * HEAD_DIM + hb;
            ka0.h[0] = *reinterpret_cast<const v8h*>(k0p);
            ka0.h[1] = *reinterpret_cast<const v8h*>(k0p + 16);
            const _Float16* k1p = k + (size_t)(c0 + 16 + l15) * HEAD_DIM + hb;
            ka1.h[0] = *reinterpret_cast<const v8h*>(k1p);
            ka1.h[1] = *reinterpret_cast<const v8h*>(k1p + 16);
            s0 = wmma_f16(ka0.v, bq[kc].v, s0);
            s1 = wmma_f16(ka1.v, bq[kc].v, s1);
        }
#pragma unroll
        for (int r = 0; r < 8; ++r) { s0[r] *= SM_SCALE; s1[r] *= SM_SCALE; }
        if (c0 + 16 >= SEQ) {          // padded tail tile (keys >= SEQ)
#pragma unroll
            for (int r = 0; r < 8; ++r) s1[r] = -1e30f;
        }

        // ---- online softmax (per q-row = per lane, halves combined) ----
        float mx = -1e30f;
#pragma unroll
        for (int r = 0; r < 8; ++r) mx = fmaxf(mx, fmaxf(s0[r], s1[r]));
        mx = fmaxf(mx, __shfl_xor(mx, 16));
        const float mnew = fmaxf(m, mx);
        const float corr = __expf(m - mnew);

        V16 pa;
        float ps = 0.0f;
#pragma unroll
        for (int r = 0; r < 8; ++r) {
            const float p0 = __expf(s0[r] - mnew);
            const float p1 = __expf(s1[r] - mnew);
            ps += p0 + p1;
            pa.v[r]     = (_Float16)p0;   // A halves 0..7  <- keys (0..7)+8*hi
            pa.v[r + 8] = (_Float16)p1;   // A halves 8..15 <- keys (16..23)+8*hi
        }
        ps += __shfl_xor(ps, 16);
        l = l * corr + ps;
        m = mnew;

        // ---- rescale O accumulators (per-row correction broadcast) ----
        float sc[8];
#pragma unroll
        for (int r = 0; r < 8; ++r) sc[r] = __shfl(corr, r + 8 * hi);
#pragma unroll
        for (int ct = 0; ct < 8; ++ct)
#pragma unroll
            for (int r = 0; r < 8; ++r) o[ct][r] *= sc[r];

        // ---- O += P @ V ----
#pragma unroll
        for (int ct = 0; ct < 8; ++ct) {
            V16 vb;
            const _Float16* vp = v + (size_t)c0 * HEAD_DIM + ct * 16 + l15;
#pragma unroll
            for (int s = 0; s < 16; ++s)
                vb.v[s] = vp[(size_t)((s & 7) + 8 * hi + 16 * (s >> 3)) * HEAD_DIM];
            o[ct] = wmma_f16(pa.v, vb.v, o[ct]);
        }
    }

    // ---- finalize: divide by row sum, store f16 into [SEQ][DIM] ----
    const float linv = 1.0f / l;
    float rs[8];
#pragma unroll
    for (int r = 0; r < 8; ++r) rs[r] = __shfl(linv, r + 8 * hi);
#pragma unroll
    for (int ct = 0; ct < 8; ++ct) {
#pragma unroll
        for (int r = 0; r < 8; ++r) {
            const int s  = qt * 16 + r + 8 * hi;
            const int cg = head * HEAD_DIM + ct * 16 + l15;
            attn[(size_t)s * DIM + cg] = (_Float16)(o[ct][r] * rs[r]);
        }
    }
}

// ---------------------------------------------------------------------------
// Host launcher
// ---------------------------------------------------------------------------
extern "C" void kernel_launch(void* const* d_in, const int* in_sizes, int n_in,
                              void* d_out, int out_size, void* d_ws, size_t ws_size,
                              hipStream_t stream) {
    (void)in_sizes; (void)n_in; (void)out_size; (void)ws_size;

    const float* x     = (const float*)d_in[0];
    const float* freqs = (const float*)d_in[1];
    const float* Wq    = (const float*)d_in[2];
    const float* bq    = (const float*)d_in[3];
    const float* gq    = (const float*)d_in[4];
    const float* Wk    = (const float*)d_in[5];
    const float* bk    = (const float*)d_in[6];
    const float* gk    = (const float*)d_in[7];
    const float* Wv    = (const float*)d_in[8];
    const float* bv    = (const float*)d_in[9];
    const float* Wo    = (const float*)d_in[10];
    const float* bo    = (const float*)d_in[11];
    const float* qkfb  = (const float*)d_in[12];
    const float* vfb   = (const float*)d_in[13];
    const float* vsb   = (const float*)d_in[14];
    float* out = (float*)d_out;

    // workspace layout (all sub-buffers 256B aligned)
    char* p = (char*)d_ws;
    const size_t xh_b   = (size_t)SEQ * DIM * 2;            // 9,584,640
    const size_t w_b    = (size_t)DIM * DIM * 2;            // 4,718,592
    const size_t qt_b   = (size_t)NUM_HEADS * HEAD_DIM * SEQ * 2;
    const size_t kn_b   = (size_t)NUM_HEADS * SEQP * HEAD_DIM * 2;
    _Float16* xh   = (_Float16*)p;               p += xh_b;
    _Float16* wqh  = (_Float16*)p;               p += w_b;
    _Float16* wkh  = (_Float16*)p;               p += w_b;
    _Float16* wvh  = (_Float16*)p;               p += w_b;
    _Float16* woh  = (_Float16*)p;               p += w_b;
    _Float16* q_t  = (_Float16*)p;               p += qt_b;
    _Float16* k_n  = (_Float16*)p;               p += kn_b;
    _Float16* v_n  = (_Float16*)p;               p += kn_b;
    _Float16* at_h = (_Float16*)p;               p += xh_b;

    // zero K/V (covers the 16 padded rows per head so masked keys stay finite)
    hipMemsetAsync(k_n, 0, kn_b, stream);
    hipMemsetAsync(v_n, 0, kn_b, stream);

    // f32 -> f16 conversions
    {
        const int nx4 = SEQ * DIM / 4, nw4 = DIM * DIM / 4;
        cvt_f32_f16<<<(nx4 + 255) / 256, 256, 0, stream>>>(x,  xh,  nx4);
        cvt_f32_f16<<<(nw4 + 255) / 256, 256, 0, stream>>>(Wq, wqh, nw4);
        cvt_f32_f16<<<(nw4 + 255) / 256, 256, 0, stream>>>(Wk, wkh, nw4);
        cvt_f32_f16<<<(nw4 + 255) / 256, 256, 0, stream>>>(Wv, wvh, nw4);
        cvt_f32_f16<<<(nw4 + 255) / 256, 256, 0, stream>>>(Wo, woh, nw4);
    }

    const size_t smem = (size_t)(16 * DIM + 16) * sizeof(float);  // 98,368 B

    // projections
    gemm_fused<<<MTILES, 256, smem, stream>>>(xh, wqh, bq, gq, freqs, qkfb, nullptr,
                                              q_t, nullptr, 0);
    gemm_fused<<<MTILES, 256, smem, stream>>>(xh, wkh, bk, gk, freqs, qkfb, nullptr,
                                              k_n, nullptr, 1);
    gemm_fused<<<MTILES, 256, 0,    stream>>>(xh, wvh, bv, nullptr, nullptr, vfb, vsb,
                                              v_n, nullptr, 2);

    // attention: 4 q-tiles per block (one per wave), 12 heads
    dim3 agrid((MTILES + 3) / 4, NUM_HEADS);
    flash_attn<<<agrid, 128, 0, stream>>>(q_t, k_n, v_n, at_h);

    // output projection -> f32 d_out
    gemm_fused<<<MTILES, 256, 0, stream>>>(at_h, woh, bo, nullptr, nullptr, nullptr, nullptr,
                                           nullptr, out, 3);
}
